// RelativeAttention_5652176961770
// MI455X (gfx1250) — matrix operand order repaired and checked by reading, compiled, not run on hardware
//
#include <hip/hip_runtime.h>
#include <hip/hip_bf16.h>

// ---------------- problem constants ----------------
#define T_   2049
#define D_   1024
#define H_   16
#define HD_  64
#define TP   2064            // T padded to 16 (129 tiles)
#define TQ   2080            // qh rows padded further (need row t+1 up to 2064)
#define NTM  129             // TP/16
#define NCB  16              // D/64 column blocks (one head wide)
#define WPB  4               // waves per block

typedef __bf16 bf16;
typedef __attribute__((ext_vector_type(16))) __bf16 v16bf;
typedef __attribute__((ext_vector_type(8)))  __bf16 v8bf;
typedef __attribute__((ext_vector_type(8)))  float  v8f;

#define DEVINL __device__ __forceinline__

// ---- WMMA fragment loaders (CDNA5 ISA 7.12.2 layouts, wave32) ----
// A (16x32 bf16): lane l(0..15) holds row M=l; halves j: K = 8*hi + (j&7) + 16*(j>>3)
DEVINL v16bf load_a_frag(const bf16* base, int ld, int row0, int k0, int lane) {
    int l = lane & 15, hi = lane >> 4;
    const bf16* r = base + (size_t)(row0 + l) * (size_t)ld + k0;
    v8bf lo = *(const v8bf*)(r + 8 * hi);
    v8bf hh = *(const v8bf*)(r + 16 + 8 * hi);
    v16bf a;
#pragma unroll
    for (int j = 0; j < 8; ++j) { a[j] = lo[j]; a[j + 8] = hh[j]; }
    return a;
}

// B (32x16 bf16): lane l holds column N=l; halves j: K = k0 + 8*hi + (j&7) + 16*(j>>3).
// base points at element (k0, n0); element (k,n) lives at base + n*stride + (k-k0)
DEVINL v16bf load_b_frag(const bf16* base, size_t stride, int lane) {
    int l = lane & 15, hi = lane >> 4;
    const bf16* r = base + (size_t)l * stride;
    v8bf lo = *(const v8bf*)(r + 8 * hi);
    v8bf hh = *(const v8bf*)(r + 16 + 8 * hi);
    v16bf b;
#pragma unroll
    for (int j = 0; j < 8; ++j) { b[j] = lo[j]; b[j + 8] = hh[j]; }
    return b;
}

DEVINL v8f wmma_bf16(v16bf a, v16bf b, v8f c) {
    return __builtin_amdgcn_wmma_f32_16x16x32_bf16(false, a, false, b, (short)0, c,
                                                   false, false);
}

// ---------------- utility kernels ----------------
__global__ void zero_u32_kernel(unsigned int* dst, long long n) {
    long long i = (long long)blockIdx.x * blockDim.x + threadIdx.x;
    long long stride = (long long)gridDim.x * blockDim.x;
    for (; i < n; i += stride) dst[i] = 0u;
}

__global__ void cvt_f32_bf16_kernel(const float* __restrict__ src, bf16* __restrict__ dst,
                                    long long n) {
    long long i = (long long)blockIdx.x * blockDim.x + threadIdx.x;
    long long stride = (long long)gridDim.x * blockDim.x;
    for (; i < n; i += stride) dst[i] = (bf16)src[i];
}

// ---------------- projection: out = X @ W^T + b ----------------
// One wave computes 16 rows x 64 cols (= one head's width), software-pipelined:
// fragments for step k+32 are loaded (distinct live ranges) before the step-k
// WMMAs execute, so loads overlap the 4-WMMA group.
// X: (TP, D) bf16 row-major.  W: (D, D) bf16 row-major (element W[o][d]).
// mode 0: Q -> qh (H, TQ, HD);  1: K -> kh (H, TP, HD);  2: V -> vt (H, HD, TP)
__global__ __launch_bounds__(32 * WPB) void proj_kernel(
    const bf16* __restrict__ X, const bf16* __restrict__ W,
    const float* __restrict__ bias, bf16* __restrict__ out, int mode) {
    int lane = threadIdx.x & 31;
    int wid = blockIdx.x * WPB + (threadIdx.x >> 5);
    if (wid >= NTM * NCB) return;
    int tm = wid / NCB, cb = wid % NCB;
    int row0 = tm * 16, col0 = cb * 64;

    const bf16* w0 = W + (size_t)(col0 +  0) * D_;
    const bf16* w1 = W + (size_t)(col0 + 16) * D_;
    const bf16* w2 = W + (size_t)(col0 + 32) * D_;
    const bf16* w3 = W + (size_t)(col0 + 48) * D_;

    v8f a0 = {}, a1 = {}, a2 = {}, a3 = {};
    v16bf a  = load_a_frag(X, D_, row0, 0, lane);
    v16bf b0 = load_b_frag(w0, (size_t)D_, lane);
    v16bf b1 = load_b_frag(w1, (size_t)D_, lane);
    v16bf b2 = load_b_frag(w2, (size_t)D_, lane);
    v16bf b3 = load_b_frag(w3, (size_t)D_, lane);
    for (int kk = 0; kk < D_; kk += 32) {
        int kn = (kk + 32 < D_) ? kk + 32 : kk;   // last iter: harmless reload
        v16bf an  = load_a_frag(X, D_, row0, kn, lane);
        v16bf c0  = load_b_frag(w0 + kn, (size_t)D_, lane);
        v16bf c1  = load_b_frag(w1 + kn, (size_t)D_, lane);
        v16bf c2  = load_b_frag(w2 + kn, (size_t)D_, lane);
        v16bf c3  = load_b_frag(w3 + kn, (size_t)D_, lane);
        a0 = wmma_bf16(a, b0, a0);
        a1 = wmma_bf16(a, b1, a1);
        a2 = wmma_bf16(a, b2, a2);
        a3 = wmma_bf16(a, b3, a3);
        a = an; b0 = c0; b1 = c1; b2 = c2; b3 = c3;
    }
    int l = lane & 15, hi = lane >> 4;
    int h = cb;                        // 64-wide block == head
    float bv0 = bias[col0 + l];
    float bv1 = bias[col0 + 16 + l];
    float bv2 = bias[col0 + 32 + l];
    float bv3 = bias[col0 + 48 + l];
#pragma unroll
    for (int r = 0; r < 8; ++r) {
        int t = row0 + (hi << 3) + r;
        float v0 = a0[r] + bv0, v1 = a1[r] + bv1, v2 = a2[r] + bv2, v3 = a3[r] + bv3;
        if (mode == 0) {
            bf16* o = out + ((size_t)h * TQ + t) * HD_;
            o[l] = (bf16)v0; o[16 + l] = (bf16)v1; o[32 + l] = (bf16)v2; o[48 + l] = (bf16)v3;
        } else if (mode == 1) {
            bf16* o = out + ((size_t)h * TP + t) * HD_;
            o[l] = (bf16)v0; o[16 + l] = (bf16)v1; o[32 + l] = (bf16)v2; o[48 + l] = (bf16)v3;
        } else {
            bf16* o = out + (size_t)h * HD_ * TP + t;
            o[(size_t)(l)      * TP] = (bf16)v0;
            o[(size_t)(16 + l) * TP] = (bf16)v1;
            o[(size_t)(32 + l) * TP] = (bf16)v2;
            o[(size_t)(48 + l) * TP] = (bf16)v3;
        }
    }
}

// ---------------- fused flash attention with relative shift ----------------
// scores[t,s] = qh[t].kh[s] + qh[t+1].Er[s]  (accumulated into one C fragment);
// mask s<=t; softmax(/8); O = P @ V.  All 12 B fragments of a key tile are
// loaded up-front so the V fragments land during the softmax VALU section.
__global__ __launch_bounds__(32 * WPB) void attn_kernel(
    const bf16* __restrict__ qh, const bf16* __restrict__ kh,
    const bf16* __restrict__ vt, const bf16* __restrict__ er,
    bf16* __restrict__ ctx) {
    __shared__ __align__(16) bf16 pbuf[WPB][16 * 32];
    int lane = threadIdx.x & 31;
    int wib = threadIdx.x >> 5;
    int wid = blockIdx.x * WPB + wib;
    if (wid >= H_ * NTM) return;
    int h = wid / NTM, qi = wid % NTM;
    int row0 = qi * 16;
    int l = lane & 15, hi = lane >> 4;

    const bf16* qhh = qh + (size_t)h * TQ * HD_;
    const bf16* khh = kh + (size_t)h * TP * HD_;
    const bf16* vth = vt + (size_t)h * HD_ * TP;
    bf16* pb = pbuf[wib];

    // Q and shifted-Q fragments (K = 0..31 and 32..63)
    v16bf qa0 = load_a_frag(qhh, HD_, row0, 0, lane);
    v16bf qa1 = load_a_frag(qhh, HD_, row0, 32, lane);
    v16bf qs0 = load_a_frag(qhh, HD_, row0 + 1, 0, lane);
    v16bf qs1 = load_a_frag(qhh, HD_, row0 + 1, 32, lane);

    float rowmax[8], rowsum[8];
#pragma unroll
    for (int r = 0; r < 8; ++r) { rowmax[r] = -3e38f; rowsum[r] = 0.0f; }
    v8f o0 = {}, o1 = {}, o2 = {}, o3 = {};

    int nkt = qi / 2 + 1;  // 32-wide key tiles covering s <= row0+15
    for (int kt = 0; kt < nkt; ++kt) {
        int s0 = kt * 32;
        // ---- all fragment loads for this tile issued up-front ----
        v16bf kb00 = load_b_frag(khh + (size_t)s0 * HD_,             (size_t)HD_, lane);
        v16bf kb01 = load_b_frag(khh + (size_t)s0 * HD_ + 32,        (size_t)HD_, lane);
        v16bf kb10 = load_b_frag(khh + (size_t)(s0 + 16) * HD_,      (size_t)HD_, lane);
        v16bf kb11 = load_b_frag(khh + (size_t)(s0 + 16) * HD_ + 32, (size_t)HD_, lane);
        v16bf eb00 = load_b_frag(er + (size_t)s0 * HD_,              (size_t)HD_, lane);
        v16bf eb01 = load_b_frag(er + (size_t)s0 * HD_ + 32,         (size_t)HD_, lane);
        v16bf eb10 = load_b_frag(er + (size_t)(s0 + 16) * HD_,       (size_t)HD_, lane);
        v16bf eb11 = load_b_frag(er + (size_t)(s0 + 16) * HD_ + 32,  (size_t)HD_, lane);
        v16bf vb0  = load_b_frag(vth + s0,                   (size_t)TP, lane);
        v16bf vb1  = load_b_frag(vth + (size_t)16 * TP + s0, (size_t)TP, lane);
        v16bf vb2  = load_b_frag(vth + (size_t)32 * TP + s0, (size_t)TP, lane);
        v16bf vb3  = load_b_frag(vth + (size_t)48 * TP + s0, (size_t)TP, lane);

        // ---- scores + relative term: two independent 4-WMMA chains ----
        v8f S0 = {}, S1 = {};
        S0 = wmma_bf16(qa0, kb00, S0);
        S1 = wmma_bf16(qa0, kb10, S1);
        S0 = wmma_bf16(qa1, kb01, S0);
        S1 = wmma_bf16(qa1, kb11, S1);
        S0 = wmma_bf16(qs0, eb00, S0);
        S1 = wmma_bf16(qs0, eb10, S1);
        S0 = wmma_bf16(qs1, eb01, S0);
        S1 = wmma_bf16(qs1, eb11, S1);

        // ---- online softmax (row stats across 16 lanes of each half) ----
#pragma unroll
        for (int r = 0; r < 8; ++r) {
            int t = row0 + (hi << 3) + r;
            float l0 = S0[r] * 0.125f;
            float l1 = S1[r] * 0.125f;
            if (s0 + l > t)      l0 = -1e30f;
            if (s0 + 16 + l > t) l1 = -1e30f;
            float m = fmaxf(l0, l1);
#pragma unroll
            for (int off = 8; off; off >>= 1) m = fmaxf(m, __shfl_xor(m, off, 16));
            m = fmaxf(m, rowmax[r]);
            float alpha = __expf(rowmax[r] - m);
            float p0 = __expf(l0 - m), p1 = __expf(l1 - m);
            float ps = p0 + p1;
#pragma unroll
            for (int off = 8; off; off >>= 1) ps += __shfl_xor(ps, off, 16);
            rowsum[r] = rowsum[r] * alpha + ps;
            rowmax[r] = m;
            o0[r] *= alpha; o1[r] *= alpha; o2[r] *= alpha; o3[r] *= alpha;
            int mrow = (hi << 3) + r;
            pb[mrow * 32 + l]      = (bf16)p0;
            pb[mrow * 32 + 16 + l] = (bf16)p1;
        }
        asm volatile("s_wait_dscnt 0" ::: "memory");
        v16bf pa = load_a_frag(pb, 32, 0, 0, lane);   // P: 16x32, A layout
        o0 = wmma_bf16(pa, vb0, o0);
        o1 = wmma_bf16(pa, vb1, o1);
        o2 = wmma_bf16(pa, vb2, o2);
        o3 = wmma_bf16(pa, vb3, o3);
    }

#pragma unroll
    for (int r = 0; r < 8; ++r) {
        int t = row0 + (hi << 3) + r;
        if (t < T_) {
            float inv = 1.0f / rowsum[r];
            size_t base = (size_t)t * D_ + (size_t)h * HD_;
            ctx[base + l]      = (bf16)(o0[r] * inv);
            ctx[base + 16 + l] = (bf16)(o1[r] * inv);
            ctx[base + 32 + l] = (bf16)(o2[r] * inv);
            ctx[base + 48 + l] = (bf16)(o3[r] * inv);
        }
    }
}

// ---------------- output projection: out = ctx @ Wo^T + b (f32 out) ----------------
// Same software-pipelined 16x64 blocking as proj_kernel.
__global__ __launch_bounds__(32 * WPB) void oproj_kernel(
    const bf16* __restrict__ X, const bf16* __restrict__ W,
    const float* __restrict__ bias, float* __restrict__ out) {
    int lane = threadIdx.x & 31;
    int wid = blockIdx.x * WPB + (threadIdx.x >> 5);
    if (wid >= NTM * NCB) return;
    int tm = wid / NCB, cb = wid % NCB;
    int row0 = tm * 16, col0 = cb * 64;

    const bf16* w0 = W + (size_t)(col0 +  0) * D_;
    const bf16* w1 = W + (size_t)(col0 + 16) * D_;
    const bf16* w2 = W + (size_t)(col0 + 32) * D_;
    const bf16* w3 = W + (size_t)(col0 + 48) * D_;

    v8f a0 = {}, a1 = {}, a2 = {}, a3 = {};
    v16bf a  = load_a_frag(X, D_, row0, 0, lane);
    v16bf b0 = load_b_frag(w0, (size_t)D_, lane);
    v16bf b1 = load_b_frag(w1, (size_t)D_, lane);
    v16bf b2 = load_b_frag(w2, (size_t)D_, lane);
    v16bf b3 = load_b_frag(w3, (size_t)D_, lane);
    for (int kk = 0; kk < D_; kk += 32) {
        int kn = (kk + 32 < D_) ? kk + 32 : kk;
        v16bf an = load_a_frag(X, D_, row0, kn, lane);
        v16bf c0 = load_b_frag(w0 + kn, (size_t)D_, lane);
        v16bf c1 = load_b_frag(w1 + kn, (size_t)D_, lane);
        v16bf c2 = load_b_frag(w2 + kn, (size_t)D_, lane);
        v16bf c3 = load_b_frag(w3 + kn, (size_t)D_, lane);
        a0 = wmma_bf16(a, b0, a0);
        a1 = wmma_bf16(a, b1, a1);
        a2 = wmma_bf16(a, b2, a2);
        a3 = wmma_bf16(a, b3, a3);
        a = an; b0 = c0; b1 = c1; b2 = c2; b3 = c3;
    }
    int l = lane & 15, hi = lane >> 4;
    float bv0 = bias[col0 + l];
    float bv1 = bias[col0 + 16 + l];
    float bv2 = bias[col0 + 32 + l];
    float bv3 = bias[col0 + 48 + l];
#pragma unroll
    for (int r = 0; r < 8; ++r) {
        int t = row0 + (hi << 3) + r;
        if (t < T_) {
            float* o = out + (size_t)t * D_ + col0;
            o[l]      = a0[r] + bv0;
            o[16 + l] = a1[r] + bv1;
            o[32 + l] = a2[r] + bv2;
            o[48 + l] = a3[r] + bv3;
        }
    }
}

// ---------------- host launch ----------------
extern "C" void kernel_launch(void* const* d_in, const int* in_sizes, int n_in,
                              void* d_out, int out_size, void* d_ws, size_t ws_size,
                              hipStream_t stream) {
    const float* q    = (const float*)d_in[0];
    const float* k    = (const float*)d_in[1];
    const float* v    = (const float*)d_in[2];
    // d_in[3] = mask (causal tril) -- implemented analytically
    const float* Wq_w = (const float*)d_in[4];
    const float* Wq_b = (const float*)d_in[5];
    const float* Wk_w = (const float*)d_in[6];
    const float* Wk_b = (const float*)d_in[7];
    const float* Wv_w = (const float*)d_in[8];
    const float* Wv_b = (const float*)d_in[9];
    const float* Er   = (const float*)d_in[10];
    const float* Wo_w = (const float*)d_in[11];
    const float* Wo_b = (const float*)d_in[12];
    float* out = (float*)d_out;

    // workspace layout (bf16 elements); zeroed region first
    const size_t qb_e  = (size_t)TP * D_;        // padded bf16 q/k/v
    const size_t er_e  = (size_t)TP * HD_;
    const size_t qh_e  = (size_t)H_ * TQ * HD_;
    const size_t ctx_e = (size_t)TP * D_;
    const size_t w_e   = (size_t)D_ * D_;
    const size_t kh_e  = (size_t)H_ * TP * HD_;

    bf16* p    = (bf16*)d_ws;
    bf16* qb   = p; p += qb_e;
    bf16* kb   = p; p += qb_e;
    bf16* vb   = p; p += qb_e;
    bf16* erb  = p; p += er_e;
    bf16* qh   = p; p += qh_e;
    bf16* ctx  = p; p += ctx_e;
    const size_t zero_elems = 3 * qb_e + er_e + qh_e + ctx_e;
    bf16* wqb  = p; p += w_e;
    bf16* wkb  = p; p += w_e;
    bf16* wvb  = p; p += w_e;
    bf16* wob  = p; p += w_e;
    bf16* kh   = p; p += kh_e;
    bf16* vt   = p; p += kh_e;

    // 1) zero padded/partially-written buffers
    {
        long long n32 = (long long)(zero_elems * sizeof(bf16) / 4);
        zero_u32_kernel<<<2048, 256, 0, stream>>>((unsigned int*)d_ws, n32);
    }
    // 2) f32 -> bf16 conversions
    cvt_f32_bf16_kernel<<<2048, 256, 0, stream>>>(q, qb, (long long)T_ * D_);
    cvt_f32_bf16_kernel<<<2048, 256, 0, stream>>>(k, kb, (long long)T_ * D_);
    cvt_f32_bf16_kernel<<<2048, 256, 0, stream>>>(v, vb, (long long)T_ * D_);
    cvt_f32_bf16_kernel<<<512, 256, 0, stream>>>(Er, erb, (long long)T_ * HD_);
    cvt_f32_bf16_kernel<<<2048, 256, 0, stream>>>(Wq_w, wqb, (long long)D_ * D_);
    cvt_f32_bf16_kernel<<<2048, 256, 0, stream>>>(Wk_w, wkb, (long long)D_ * D_);
    cvt_f32_bf16_kernel<<<2048, 256, 0, stream>>>(Wv_w, wvb, (long long)D_ * D_);
    cvt_f32_bf16_kernel<<<2048, 256, 0, stream>>>(Wo_w, wob, (long long)D_ * D_);

    // 3) projections (one wave per 16x64 output block)
    int proj_blocks = (NTM * NCB + WPB - 1) / WPB;
    proj_kernel<<<proj_blocks, 32 * WPB, 0, stream>>>(qb, wqb, Wq_b, qh, 0);
    proj_kernel<<<proj_blocks, 32 * WPB, 0, stream>>>(kb, wkb, Wk_b, kh, 1);
    proj_kernel<<<proj_blocks, 32 * WPB, 0, stream>>>(vb, wvb, Wv_b, vt, 2);

    // 4) fused attention (one wave per head x 16-row query tile)
    int attn_blocks = (H_ * NTM + WPB - 1) / WPB;
    attn_kernel<<<attn_blocks, 32 * WPB, 0, stream>>>(qh, kh, vt, erb, ctx);

    // 5) output projection
    oproj_kernel<<<proj_blocks, 32 * WPB, 0, stream>>>(ctx, wob, Wo_b, out);

    (void)in_sizes; (void)n_in; (void)out_size; (void)ws_size;
}
